// EmotionModel_75213467288241
// MI455X (gfx1250) — matrix operand order repaired and checked by reading, not compiled, not run on hardware
//
#include <hip/hip_runtime.h>
#include <hip/hip_bf16.h>

typedef __attribute__((ext_vector_type(16))) _Float16 v16h;
typedef __attribute__((ext_vector_type(4)))  _Float16 v4h;
typedef __attribute__((ext_vector_type(2)))  _Float16 v2h;
typedef __attribute__((ext_vector_type(8)))  float    v8f;

#define T_STEPS 10
#define D_IN    512
#define HID     128
#define GATES   512   // 4*HID
#define BM      32    // batch rows per block (2 WMMA M-tiles)
#define NTILES  32    // GATES/16
#define KT_X    16    // D_IN/32
#define KT_H    4     // HID/32

#define WF_WORDS  (NTILES*KT_X*32*8)   // 131072 dwords (512KB) per 512x512 weight
#define WHH_WORDS (NTILES*KT_H*32*8)   // 32768 dwords (128KB)

static __device__ __forceinline__ unsigned int pack2h(float lo, float hi) {
    union { _Float16 h[2]; unsigned int u; } c;
    c.h[0] = (_Float16)lo; c.h[1] = (_Float16)hi;
    return c.u;
}

// Pre-pack weights into the B-fragment image used here (contiguous k, same order as loadA):
// fragment(nTile,kTile): 32 lanes x 8 dwords; lane L dword V holds
// { B[k][n], B[k+1][n] } with n = nTile*16 + (L&15), k = kTile*32 + 16*(L>=16) + 2V,
// and B[k][n] = W[n][k]  (i.e. W^T, since gates = x @ W^T).
__global__ void pack_weights_kernel(const float* __restrict__ Wih_f,
                                    const float* __restrict__ Whh_f,
                                    const float* __restrict__ Wih_b,
                                    unsigned int* __restrict__ wfF,
                                    unsigned int* __restrict__ wbF,
                                    unsigned int* __restrict__ whhF) {
    unsigned int g = blockIdx.x * blockDim.x + threadIdx.x;
    const float* W; unsigned int* dst; unsigned int w; int nTile, kTile, ldK;
    if (g < WF_WORDS)            { W = Wih_f; dst = wfF;  w = g;               nTile = w >> 12; kTile = (w >> 8) & 15; ldK = D_IN; }
    else if (g < 2u*WF_WORDS)    { W = Wih_b; dst = wbF;  w = g - WF_WORDS;    nTile = w >> 12; kTile = (w >> 8) & 15; ldK = D_IN; }
    else if (g < 2u*WF_WORDS + WHH_WORDS)
                                 { W = Whh_f; dst = whhF; w = g - 2u*WF_WORDS; nTile = w >> 10; kTile = (w >> 8) & 3;  ldK = HID;  }
    else return;
    int lane = (w >> 3) & 31;
    int v    = w & 7;
    int n = nTile * 16 + (lane & 15);
    int k = kTile * 32 + ((lane >> 4) & 1) * 16 + 2 * v;
    dst[w] = pack2h(W[n * ldK + k], W[n * ldK + k + 1]);
}

// Gather a WMMA A-fragment (16x32 f16) from an LDS tile (row-major, stride ld).
// Order used here (contiguous k, same as packed B): lane m=L&15; lanes 0-15: V0..7 -> K pairs 0..15;
// lanes 16-31 add +16 to K. Pairs are contiguous, so this lowers to 2x ds_load_b128.
static __device__ __forceinline__ v16h loadA(const _Float16* sA, int ld, int kBase, int lane) {
    int m   = lane & 15;
    int khi = ((lane >> 4) & 1) * 16;
    v16h a;
#pragma unroll
    for (int v = 0; v < 8; ++v) {
        int k = kBase + 2 * v + khi;
        v2h p = *(const v2h*)(sA + m * ld + k);
        a[2 * v]     = p.x;
        a[2 * v + 1] = p.y;
    }
    return a;
}

static __device__ __forceinline__ float sigf(float x) {
    return 1.f / (1.f + __expf(-x));
}

__global__ __launch_bounds__(256, 4)
void emotion_fused_kernel(const float* __restrict__ x,
                          const float* __restrict__ gamma,
                          const float* __restrict__ beta,
                          const unsigned int* __restrict__ wfF,
                          const unsigned int* __restrict__ wbF,
                          const unsigned int* __restrict__ whhF,
                          const float* __restrict__ bih_f, const float* __restrict__ bhh_f,
                          const float* __restrict__ bih_b, const float* __restrict__ bhh_b,
                          const float* __restrict__ W1, const float* __restrict__ b1,
                          const float* __restrict__ W2, const float* __restrict__ b2,
                          float* __restrict__ out) {
    __shared__ _Float16 sX[BM * D_IN];        // layernormed x tile (current t), f16  (32KB)
    __shared__ float    sGate[BM * GATES];    // WMMA gate pre-activations           (64KB)
    __shared__ _Float16 sH[BM * HID];         // hidden state, f16 (next-step A)      (8KB)
    __shared__ float    sLast[BM * 2 * HID];  // concat(hf, hb)                      (32KB)
    __shared__ float    sMLP[BM * 64];        //                                      (8KB)
    __shared__ float    sBias[GATES], sBiasB[GATES];

    const int tid  = threadIdx.x;
    const int lane = tid & 31;
    const int wave = tid >> 5;
    const int b0   = blockIdx.x * BM;
    const int row  = tid >> 3;   // 0..31 (one x-row per 8 threads for LN)
    const int col  = tid & 7;

    for (int i = tid; i < GATES; i += 256) {
        sBias[i]  = bih_f[i] + bhh_f[i];
        sBiasB[i] = bih_b[i] + bhh_b[i];
    }
#pragma unroll
    for (int q = 0; q < 16; ++q) sH[tid + 256 * q] = (_Float16)0.f;

    float c_reg[16];
#pragma unroll
    for (int q = 0; q < 16; ++q) c_reg[q] = 0.f;
    float hf[16];

    const int ntBase = wave * 4;   // each wave owns 4 of 32 N-tiles (and both M-tiles)
    v8f acc[8];                    // [mt*4 + nt]
    const v8f vzero = {};
    const float4* g4 = (const float4*)gamma;
    const float4* be4 = (const float4*)beta;

    for (int t = 0; t < T_STEPS; ++t) {
        // ---- LayerNorm over D=512; 8 threads/row, float4 coalesced loads ----
        {
            const float4* xr = (const float4*)(x + ((size_t)(b0 + row) * T_STEPS + t) * D_IN);
            if (t + 1 < T_STEPS) __builtin_prefetch(xr + D_IN / 4, 0, 0);
            float4 vals[16]; float s = 0.f, s2 = 0.f;
#pragma unroll
            for (int q = 0; q < 16; ++q) {
                float4 v = xr[col + 8 * q];
                vals[q] = v;
                s  += v.x + v.y + v.z + v.w;
                s2 += v.x * v.x + v.y * v.y + v.z * v.z + v.w * v.w;
            }
#pragma unroll
            for (int m = 4; m >= 1; m >>= 1) {
                s  += __shfl_xor(s,  m, 32);
                s2 += __shfl_xor(s2, m, 32);
            }
            float mu   = s * (1.f / 512.f);
            float rstd = rsqrtf(s2 * (1.f / 512.f) - mu * mu + 1e-5f);
#pragma unroll
            for (int q = 0; q < 16; ++q) {
                int c = col + 8 * q;           // float4 chunk index, 0..127
                float4 v = vals[q], g = g4[c], bb = be4[c];
                v4h hh;
                hh.x = (_Float16)((v.x - mu) * rstd * g.x + bb.x);
                hh.y = (_Float16)((v.y - mu) * rstd * g.y + bb.y);
                hh.z = (_Float16)((v.z - mu) * rstd * g.z + bb.z);
                hh.w = (_Float16)((v.w - mu) * rstd * g.w + bb.w);
                *(v4h*)(sX + row * D_IN + 4 * c) = hh;
            }
        }
        __syncthreads();

        // ---- gates = xn @ Wih_f^T + h @ Whh_f^T  (f16 WMMA, f32 accumulate)
        //      2 M-tiles per block -> each B fragment feeds 2 WMMAs.
        //      kt loops deliberately NOT unrolled: keeps exactly one iteration's
        //      fragments live (no scratch spills), cross-wave occupancy hides
        //      the load latency instead. ----
#pragma unroll
        for (int i = 0; i < 8; ++i) acc[i] = vzero;
#pragma clang loop unroll(disable)
        for (int kt = 0; kt < KT_X; ++kt) {
            v16h a0 = loadA(sX,              D_IN, kt * 32, lane);
            v16h a1 = loadA(sX + 16 * D_IN,  D_IN, kt * 32, lane);
#pragma unroll
            for (int nt = 0; nt < 4; ++nt) {
                const unsigned int* p = wfF + (((size_t)(ntBase + nt) * KT_X + kt) * 32 + lane) * 8;
                v16h bf = *(const v16h*)p;
                acc[nt]     = __builtin_amdgcn_wmma_f32_16x16x32_f16(false, a0, false, bf,
                                                                     (short)0, acc[nt],     false, false);
                acc[4 + nt] = __builtin_amdgcn_wmma_f32_16x16x32_f16(false, a1, false, bf,
                                                                     (short)0, acc[4 + nt], false, false);
            }
        }
#pragma clang loop unroll(disable)
        for (int kt = 0; kt < KT_H; ++kt) {
            v16h a0 = loadA(sH,             HID, kt * 32, lane);
            v16h a1 = loadA(sH + 16 * HID,  HID, kt * 32, lane);
#pragma unroll
            for (int nt = 0; nt < 4; ++nt) {
                const unsigned int* p = whhF + (((size_t)(ntBase + nt) * KT_H + kt) * 32 + lane) * 8;
                v16h bf = *(const v16h*)p;
                acc[nt]     = __builtin_amdgcn_wmma_f32_16x16x32_f16(false, a0, false, bf,
                                                                     (short)0, acc[nt],     false, false);
                acc[4 + nt] = __builtin_amdgcn_wmma_f32_16x16x32_f16(false, a1, false, bf,
                                                                     (short)0, acc[4 + nt], false, false);
            }
        }
        // C/D layout: lane L vgpr V holds (M = V + 8*(L>=16), N = L&15)
        {
            int mAdd = (lane >= 16) ? 8 : 0, n = lane & 15;
#pragma unroll
            for (int mt = 0; mt < 2; ++mt)
#pragma unroll
                for (int nt = 0; nt < 4; ++nt) {
                    int N0 = (ntBase + nt) * 16;
#pragma unroll
                    for (int vv = 0; vv < 8; ++vv)
                        sGate[(mt * 16 + vv + mAdd) * GATES + N0 + n] = acc[mt * 4 + nt][vv];
                }
        }
        __syncthreads();

        // ---- LSTM cell nonlinearity; c stays in registers across t ----
#pragma unroll
        for (int q = 0; q < 16; ++q) {
            int e = tid + 256 * q, m = e >> 7, j = e & 127;
            float gi = sigf(sGate[m * GATES + j]            + sBias[j]);
            float gf = sigf(sGate[m * GATES + HID + j]      + sBias[HID + j]);
            float gg = tanhf(sGate[m * GATES + 2 * HID + j] + sBias[2 * HID + j]);
            float go = sigf(sGate[m * GATES + 3 * HID + j]  + sBias[3 * HID + j]);
            c_reg[q] = gf * c_reg[q] + gi * gg;
            float h  = go * tanhf(c_reg[q]);
            sH[m * HID + j] = (_Float16)h;
            if (t == T_STEPS - 1) hf[q] = h;
        }
        __syncthreads();
    }

    // ---- backward LSTM: only its FIRST step matters (h0=c0=0 => no Whh_b term).
    //      sX still holds the layernormed t=9 tile. ----
#pragma unroll
    for (int i = 0; i < 8; ++i) acc[i] = vzero;
#pragma clang loop unroll(disable)
    for (int kt = 0; kt < KT_X; ++kt) {
        v16h a0 = loadA(sX,             D_IN, kt * 32, lane);
        v16h a1 = loadA(sX + 16 * D_IN, D_IN, kt * 32, lane);
#pragma unroll
        for (int nt = 0; nt < 4; ++nt) {
            const unsigned int* p = wbF + (((size_t)(ntBase + nt) * KT_X + kt) * 32 + lane) * 8;
            v16h bf = *(const v16h*)p;
            acc[nt]     = __builtin_amdgcn_wmma_f32_16x16x32_f16(false, a0, false, bf,
                                                                 (short)0, acc[nt],     false, false);
            acc[4 + nt] = __builtin_amdgcn_wmma_f32_16x16x32_f16(false, a1, false, bf,
                                                                 (short)0, acc[4 + nt], false, false);
        }
    }
    {
        int mAdd = (lane >= 16) ? 8 : 0, n = lane & 15;
#pragma unroll
        for (int mt = 0; mt < 2; ++mt)
#pragma unroll
            for (int nt = 0; nt < 4; ++nt) {
                int N0 = (ntBase + nt) * 16;
#pragma unroll
                for (int vv = 0; vv < 8; ++vv)
                    sGate[(mt * 16 + vv + mAdd) * GATES + N0 + n] = acc[mt * 4 + nt][vv];
            }
    }
    __syncthreads();
#pragma unroll
    for (int q = 0; q < 16; ++q) {
        int e = tid + 256 * q, m = e >> 7, j = e & 127;
        float gi = sigf(sGate[m * GATES + j]            + sBiasB[j]);
        float gg = tanhf(sGate[m * GATES + 2 * HID + j] + sBiasB[2 * HID + j]);
        float go = sigf(sGate[m * GATES + 3 * HID + j]  + sBiasB[3 * HID + j]);
        float cb = gi * gg;               // f-gate irrelevant: c0 = 0
        float hb = go * tanhf(cb);
        sLast[m * 256 + j]       = hf[q];
        sLast[m * 256 + 128 + j] = hb;
    }
    __syncthreads();

    // ---- MLP head: gelu(last @ W1^T + b1) @ W2^T + b2, tanh ----
#pragma unroll
    for (int q = 0; q < 8; ++q) {
        int e = tid + 256 * q, m = e >> 6, n = e & 63;
        float a1 = b1[n];
        const float* wrow = W1 + n * 256;
#pragma unroll 8
        for (int k = 0; k < 256; ++k) a1 += sLast[m * 256 + k] * wrow[k];
        sMLP[m * 64 + n] = 0.5f * a1 * (1.f + erff(a1 * 0.70710678118654752f));
    }
    __syncthreads();
    if (tid < BM * 5) {
        int m = tid / 5, n = tid % 5;
        float a2 = b2[n];
        const float* wrow = W2 + n * 64;
#pragma unroll 8
        for (int k = 0; k < 64; ++k) a2 += sMLP[m * 64 + k] * wrow[k];
        out[(size_t)(b0 + m) * 5 + n] = tanhf(a2);
    }
}

extern "C" void kernel_launch(void* const* d_in, const int* in_sizes, int n_in,
                              void* d_out, int out_size, void* d_ws, size_t ws_size,
                              hipStream_t stream) {
    const float* x     = (const float*)d_in[0];
    const float* gamma = (const float*)d_in[1];
    const float* beta  = (const float*)d_in[2];
    const float* Wih_f = (const float*)d_in[3];
    const float* Whh_f = (const float*)d_in[4];
    const float* bih_f = (const float*)d_in[5];
    const float* bhh_f = (const float*)d_in[6];
    const float* Wih_b = (const float*)d_in[7];
    // d_in[8] = Whh_b: provably unused (backward hidden needed only at its first
    // step where h0 == 0, so the recurrent term vanishes).
    const float* bih_b = (const float*)d_in[9];
    const float* bhh_b = (const float*)d_in[10];
    const float* W1    = (const float*)d_in[11];
    const float* b1    = (const float*)d_in[12];
    const float* W2    = (const float*)d_in[13];
    const float* b2    = (const float*)d_in[14];

    unsigned int* wfF  = (unsigned int*)d_ws;          // 512 KB
    unsigned int* wbF  = wfF + WF_WORDS;               // 512 KB
    unsigned int* whhF = wbF + WF_WORDS;               // 128 KB
    (void)ws_size; (void)n_in; (void)out_size;

    int Btot = in_sizes[0] / (T_STEPS * D_IN);

    pack_weights_kernel<<<(2 * WF_WORDS + WHH_WORDS) / 256, 256, 0, stream>>>(
        Wih_f, Whh_f, Wih_b, wfF, wbF, whhF);

    emotion_fused_kernel<<<Btot / BM, 256, 0, stream>>>(
        x, gamma, beta, wfF, wbF, whhF,
        bih_f, bhh_f, bih_b, bhh_b, W1, b1, W2, b2, (float*)d_out);
}
